// MultiHeadAttention_61933428408797
// MI455X (gfx1250) — matrix operand order repaired and checked by reading, not compiled, not run on hardware
//
#include <hip/hip_runtime.h>

typedef _Float16 half_t;
typedef __attribute__((ext_vector_type(16))) _Float16 v16h;
typedef __attribute__((ext_vector_type(8)))  _Float16 v8h;
typedef __attribute__((ext_vector_type(8)))  float    v8f;

#define DD      1024      // D_IN == D_OUT
#define NHEAD   16
#define HDIM    64
#define BATCH   2
#define SEQ     4096
#define MTOT    (BATCH*SEQ)   // 8192
#define FWAVES  4             // waves per flash block

// ---------------------------------------------------------------------------
// WMMA helpers (CDNA5 v_wmma_f32_16x16x32_f16, wave32)
// ---------------------------------------------------------------------------
__device__ __forceinline__ v8f wmma_f16(v16h a, v16h b, v8f c) {
  return __builtin_amdgcn_wmma_f32_16x16x32_f16(false, a, false, b, (short)0, c,
                                                false, false);
}

// A-matrix 16x32 f16 frag: lane%16 = row M; per-lane halves:
// j in [0,16): K = (lane>>4)*16 + j  (matches load_b_rows)
__device__ __forceinline__ v16h load_a_frag(const half_t* p, int ld) {
  const int lane = threadIdx.x & 31;
  const half_t* r = p + (size_t)(lane & 15) * ld + ((lane >> 4) << 4);
  union { v16h v; v8h h[2]; } u;
  u.h[0] = *(const v8h*)(r);
  u.h[1] = *(const v8h*)(r + 8);
  return u.v;
}

// B-matrix 32x16 frag where B[k][n] comes from source ROW n (contiguous in k):
// lane%16 = column n; halves j: K = (lane>>4)*16 + j   (contiguous 32B load)
__device__ __forceinline__ v16h load_b_rows(const half_t* p, int ld) {
  const int lane = threadIdx.x & 31;
  return *(const v16h*)(p + (size_t)(lane & 15) * ld + ((lane >> 4) << 4));
}

__device__ __forceinline__ float rmax16(float x) {
  x = fmaxf(x, __shfl_xor(x, 1, 32));
  x = fmaxf(x, __shfl_xor(x, 2, 32));
  x = fmaxf(x, __shfl_xor(x, 4, 32));
  x = fmaxf(x, __shfl_xor(x, 8, 32));
  return x;
}
__device__ __forceinline__ float rsum16(float x) {
  x += __shfl_xor(x, 1, 32);
  x += __shfl_xor(x, 2, 32);
  x += __shfl_xor(x, 4, 32);
  x += __shfl_xor(x, 8, 32);
  return x;
}

// Async-copy one 32x64 f16 tile (4KB) global -> LDS: 8 x b128, 16B per lane.
// Source rows are DD halves apart; staged tile is dense 32x64 (ld=64).
__device__ __forceinline__ void async_copy_ktile(const half_t* gsrc,
                                                 half_t* lds) {
  const int lane = threadIdx.x & 31;
  const int rowb = lane >> 3;         // 0..3
  const int colb = (lane & 7) * 16;   // byte offset within 128B row
  unsigned lbase = (unsigned)(uintptr_t)lds + (unsigned)(lane * 16);
#pragma unroll
  for (int c = 0; c < 8; ++c) {
    int row = c * 4 + rowb;
    unsigned long long g =
        (unsigned long long)(uintptr_t)((const char*)gsrc +
                                        (size_t)row * (DD * 2) + colb);
    unsigned l = lbase + (unsigned)(c * 512);
    asm volatile("global_load_async_to_lds_b128 %0, %1, off"
                 :: "v"(l), "v"(g) : "memory");
  }
}

// ---------------------------------------------------------------------------
// Elementwise prep
// ---------------------------------------------------------------------------
__global__ __launch_bounds__(256) void k_binarize(const float* __restrict__ w,
                                                  half_t* __restrict__ o, int n) {
  int i = blockIdx.x * 256 + threadIdx.x;
  if (i < n) {
    float v = w[i];
    o[i] = (half_t)((v > 0.f) ? 1.f : ((v < 0.f) ? -1.f : 0.f));
  }
}

__global__ __launch_bounds__(256) void k_cvt(const float* __restrict__ x,
                                             half_t* __restrict__ o, int n) {
  int i = blockIdx.x * 256 + threadIdx.x;
  if (i < n) o[i] = (half_t)x[i];
}

// ---------------------------------------------------------------------------
// GEMM: Y = X[M,K] * W[N,K]^T   (W rows are the B-operand columns)
// 8 waves / block (4x2); each wave computes a 32x64 tile; block tile 128x128.
// Output modes:
//   Yt != null : f16, per-head TRANSPOSED layout Yt[b*DD*SEQ + col*SEQ + s]
//   Yf != null : fp32 row-major + bias
//   else       : f16 row-major (Yh)
// ---------------------------------------------------------------------------
__global__ __launch_bounds__(256) void k_gemm_xwt(
    const half_t* __restrict__ X, const half_t* __restrict__ W,
    half_t* __restrict__ Yh, float* __restrict__ Yf,
    half_t* __restrict__ Yt, const float* __restrict__ bias,
    int M, int N, int K) {
  const int lane = threadIdx.x & 31;
  const int wv = threadIdx.x >> 5;
  const int m0 = blockIdx.x * 128 + (wv & 3) * 32;
  const int n0 = blockIdx.y * 128 + (wv >> 2) * 64;

  v8f acc[2][4];
#pragma unroll
  for (int a = 0; a < 2; ++a)
#pragma unroll
    for (int b = 0; b < 4; ++b)
#pragma unroll
      for (int r = 0; r < 8; ++r) acc[a][b][r] = 0.f;

  for (int k0 = 0; k0 < K; k0 += 32) {
    v16h a0 = load_a_frag(X + (size_t)m0 * K + k0, K);
    v16h a1 = load_a_frag(X + (size_t)(m0 + 16) * K + k0, K);
#pragma unroll
    for (int nj = 0; nj < 4; ++nj) {
      v16h bf = load_b_rows(W + (size_t)(n0 + nj * 16) * K + k0, K);
      acc[0][nj] = wmma_f16(a0, bf, acc[0][nj]);
      acc[1][nj] = wmma_f16(a1, bf, acc[1][nj]);
    }
  }

  const int cn = lane & 15, cm = (lane >> 4) * 8;
#pragma unroll
  for (int mi = 0; mi < 2; ++mi)
#pragma unroll
    for (int nj = 0; nj < 4; ++nj) {
      if (Yt) {
        // transposed store: lane owns col, 8 consecutive rows -> one v8h store
        int row0 = m0 + mi * 16 + cm;           // 8-aligned, never crosses SEQ
        int col = n0 + nj * 16 + cn;
        int b = row0 / SEQ, s = row0 % SEQ;
        union { v8h v; half_t h[8]; } u;
#pragma unroll
        for (int r = 0; r < 8; ++r) u.h[r] = (half_t)acc[mi][nj][r];
        *(v8h*)(Yt + (size_t)b * DD * SEQ + (size_t)col * SEQ + s) = u.v;
      } else {
#pragma unroll
        for (int r = 0; r < 8; ++r) {
          size_t row = (size_t)(m0 + mi * 16 + cm + r);
          int col = n0 + nj * 16 + cn;
          float v = acc[mi][nj][r];
          if (Yf)
            Yf[row * N + col] = v + (bias ? bias[col] : 0.f);
          else
            Yh[row * N + col] = (half_t)v;
        }
      }
    }
}

// ---------------------------------------------------------------------------
// Flash attention (causal), one wave per 32-query block, 4 waves/block.
// Q,K,CTX are [MTOT, DD] f16 (col = h*HDIM + d); Vt is per-head transposed:
// Vt[b*DD*SEQ + (h*HDIM+d)*SEQ + s].
// K-tiles are staged into LDS with double-buffered async copies (ASYNCcnt).
// ---------------------------------------------------------------------------
__global__ __launch_bounds__(128) void k_flash(const half_t* __restrict__ Q,
                                               const half_t* __restrict__ Km,
                                               const half_t* __restrict__ Vt,
                                               half_t* __restrict__ CTX) {
  __shared__ __attribute__((aligned(16))) half_t pbuf[FWAVES][32 * 40];
  __shared__ __attribute__((aligned(16))) half_t kstage[FWAVES][2][32 * 64];
  const int lane = threadIdx.x & 31;
  const int w = threadIdx.x >> 5;
  const int b = blockIdx.y / NHEAD, h = blockIdx.y % NHEAD;
  const int qb = (blockIdx.x * FWAVES + w) * 32;  // query row offset

  const size_t base = (size_t)b * SEQ * DD + (size_t)h * HDIM;
  const half_t* Qb = Q + base + (size_t)qb * DD;
  const half_t* Kb = Km + base;
  const half_t* Vtb = Vt + (size_t)b * DD * SEQ + (size_t)(h * HDIM) * SEQ;

  v16h aq[2][2];  // resident Q A-frags: [m-tile][k-chunk]
#pragma unroll
  for (int mi = 0; mi < 2; ++mi)
#pragma unroll
    for (int kc = 0; kc < 2; ++kc)
      aq[mi][kc] = load_a_frag(Qb + (size_t)mi * 16 * DD + kc * 32, DD);

  v8f o[2][4];
  float mr[2][8], lr[2][8];
#pragma unroll
  for (int mi = 0; mi < 2; ++mi)
#pragma unroll
    for (int r = 0; r < 8; ++r) {
      mr[mi][r] = -3.0e38f;
      lr[mi][r] = 0.f;
#pragma unroll
      for (int nd = 0; nd < 4; ++nd) o[mi][nd][r] = 0.f;
    }

  const int nkb = qb / 32 + 1;  // causal: key blocks at/below the diagonal
  async_copy_ktile(Kb, kstage[w][0]);  // prefetch first K-tile

  for (int kb = 0; kb < nkb; ++kb) {
    const half_t* kst = kstage[w][kb & 1];
    const bool pf = (kb + 1 < nkb);
    // WAR fence: last iteration's ds reads of the buffer we now overwrite
    asm volatile("s_wait_dscnt 0" ::: "memory");
    if (pf) {
      async_copy_ktile(Kb + (size_t)(kb + 1) * 32 * DD, kstage[w][(kb + 1) & 1]);
      // async loads complete in order: <=8 outstanding => current tile ready
      asm volatile("s_wait_asynccnt 0x8" ::: "memory");
    } else {
      asm volatile("s_wait_asynccnt 0x0" ::: "memory");
    }

    // ---- S = Q K^T (32x32 block), B-frags from staged LDS tile ----
    v8f s[2][2];
#pragma unroll
    for (int mi = 0; mi < 2; ++mi)
#pragma unroll
      for (int nj = 0; nj < 2; ++nj)
#pragma unroll
        for (int r = 0; r < 8; ++r) s[mi][nj][r] = 0.f;

#pragma unroll
    for (int nj = 0; nj < 2; ++nj)
#pragma unroll
      for (int kc = 0; kc < 2; ++kc) {
        v16h bk = load_b_rows(kst + (nj * 16) * 64 + kc * 32, 64);
        s[0][nj] = wmma_f16(aq[0][kc], bk, s[0][nj]);
        s[1][nj] = wmma_f16(aq[1][kc], bk, s[1][nj]);
      }

    // ---- scale + causal mask (diagonal block only) ----
    const bool diag = (kb == nkb - 1);
    const float sc = 0.125f;  // 1/sqrt(64)
#pragma unroll
    for (int mi = 0; mi < 2; ++mi)
#pragma unroll
      for (int r = 0; r < 8; ++r) {
        int qrow = qb + mi * 16 + (lane >> 4) * 8 + r;
#pragma unroll
        for (int nj = 0; nj < 2; ++nj) {
          float v = s[mi][nj][r] * sc;
          int kcol = kb * 32 + nj * 16 + (lane & 15);
          if (diag && kcol > qrow) v = -3.0e38f;
          s[mi][nj][r] = v;
        }
      }

    // ---- online softmax (row reductions across 16-lane halves) ----
#pragma unroll
    for (int mi = 0; mi < 2; ++mi)
#pragma unroll
      for (int r = 0; r < 8; ++r) {
        float bm = rmax16(fmaxf(s[mi][0][r], s[mi][1][r]));
        float mn = fmaxf(mr[mi][r], bm);
        float alpha = __expf(mr[mi][r] - mn);
        mr[mi][r] = mn;
        float p0 = __expf(s[mi][0][r] - mn);
        float p1 = __expf(s[mi][1][r] - mn);
        s[mi][0][r] = p0;
        s[mi][1][r] = p1;
        lr[mi][r] = lr[mi][r] * alpha + rsum16(p0 + p1);
#pragma unroll
        for (int nd = 0; nd < 4; ++nd) o[mi][nd][r] *= alpha;
      }

    // ---- relayout P: C-frag -> LDS (f16, row-major, stride 40) -> A-frag ----
    half_t* pb = pbuf[w];
#pragma unroll
    for (int mi = 0; mi < 2; ++mi)
#pragma unroll
      for (int nj = 0; nj < 2; ++nj)
#pragma unroll
        for (int r = 0; r < 8; ++r) {
          int row = mi * 16 + (lane >> 4) * 8 + r;
          int col = nj * 16 + (lane & 15);
          pb[row * 40 + col] = (half_t)s[mi][nj][r];
        }
    asm volatile("s_wait_dscnt 0" ::: "memory");  // same-wave LDS RAW fence
    v16h ap0 = load_a_frag(pb, 40);
    v16h ap1 = load_a_frag(pb + 16 * 40, 40);

    // ---- O += P V  (B-frags are contiguous rows of transposed V) ----
#pragma unroll
    for (int nd = 0; nd < 4; ++nd) {
      v16h bv = load_b_rows(Vtb + (size_t)(nd * 16) * SEQ + kb * 32, SEQ);
      o[0][nd] = wmma_f16(ap0, bv, o[0][nd]);
      o[1][nd] = wmma_f16(ap1, bv, o[1][nd]);
    }
  }

  // ---- epilogue: O / l -> CTX (f16) ----
  half_t* Cb = CTX + base + (size_t)qb * DD;
  const int cn = lane & 15, cm = (lane >> 4) * 8;
#pragma unroll
  for (int mi = 0; mi < 2; ++mi)
#pragma unroll
    for (int nd = 0; nd < 4; ++nd)
#pragma unroll
      for (int r = 0; r < 8; ++r) {
        float v = o[mi][nd][r] / lr[mi][r];
        Cb[(size_t)(mi * 16 + cm + r) * DD + nd * 16 + cn] = (half_t)v;
      }
}

// ---------------------------------------------------------------------------
// Launch
// ---------------------------------------------------------------------------
extern "C" void kernel_launch(void* const* d_in, const int* in_sizes, int n_in,
                              void* d_out, int out_size, void* d_ws, size_t ws_size,
                              hipStream_t stream) {
  const float* x = (const float*)d_in[0];
  const float* Wq = (const float*)d_in[1];
  const float* Wk = (const float*)d_in[2];
  const float* Wv = (const float*)d_in[3];
  const float* Wo = (const float*)d_in[4];
  const float* bo = (const float*)d_in[5];
  float* out = (float*)d_out;

  const size_t MB = 1024ull * 1024ull;
  char* ws = (char*)d_ws;
  half_t* xh = (half_t*)ws;                    // 16 MB  : x in f16
  half_t* wqb = (half_t*)(ws + 16 * MB);       // 2 MB   : sign(Wq) f16
  half_t* wkb = wqb + DD * DD;
  half_t* wvb = wkb + DD * DD;
  half_t* wob = wvb + DD * DD;
  half_t* Qh = (half_t*)(ws + 24 * MB);        // 16 MB each
  half_t* Kh = Qh + (size_t)MTOT * DD;
  half_t* Vth = Kh + (size_t)MTOT * DD;        // V in per-head transposed layout
  half_t* ctx = Vth + (size_t)MTOT * DD;       // ends at 88 MB

  const int NW = DD * DD;  // 1M weight elems
  k_cvt<<<(MTOT * DD) / 256, 256, 0, stream>>>(x, xh, MTOT * DD);
  k_binarize<<<NW / 256, 256, 0, stream>>>(Wq, wqb, NW);
  k_binarize<<<NW / 256, 256, 0, stream>>>(Wk, wkb, NW);
  k_binarize<<<NW / 256, 256, 0, stream>>>(Wv, wvb, NW);
  k_binarize<<<NW / 256, 256, 0, stream>>>(Wo, wob, NW);

  dim3 gg(MTOT / 128, DD / 128);
  k_gemm_xwt<<<gg, 256, 0, stream>>>(xh, wqb, Qh, nullptr, nullptr, nullptr,
                                     MTOT, DD, DD);
  k_gemm_xwt<<<gg, 256, 0, stream>>>(xh, wkb, Kh, nullptr, nullptr, nullptr,
                                     MTOT, DD, DD);
  k_gemm_xwt<<<gg, 256, 0, stream>>>(xh, wvb, nullptr, nullptr, Vth, nullptr,
                                     MTOT, DD, DD);

  dim3 gf(SEQ / (32 * FWAVES), BATCH * NHEAD);
  k_flash<<<gf, 32 * FWAVES, 0, stream>>>(Qh, Kh, Vth, ctx);

  k_gemm_xwt<<<gg, 256, 0, stream>>>(ctx, wob, nullptr, out, nullptr, bo,
                                     MTOT, DD, DD);
}
